// AGRUCell_6700148981889
// MI455X (gfx1250) — matrix operand order repaired and checked by reading, not compiled, not run on hardware
//
#include <hip/hip_runtime.h>

// AGRU cell, fully fused single-pass kernel for gfx1250 (MI455X).
// B=2048, T=200, D=U=64.  Grid = B/16 blocks of 128 threads (4 wave32).
// Each block owns one 16-row batch tile; wave w owns the 16-wide N-slice
// [16w, 16w+16).  GEMMs on v_wmma_f32_16x16x32_f16 with f32 accumulation;
// h kept as f32 master in registers; gate transcendentals on the HW TRANS
// pipe (v_exp_f32 / v_rcp_f32 / v_tanh_f32), branch-free.
// T is a template parameter so output-row strides fold into store IOFFSETs.

typedef __attribute__((ext_vector_type(16))) _Float16 v16h;
typedef __attribute__((ext_vector_type(8)))  _Float16 v8h;
typedef __attribute__((ext_vector_type(8)))  float    v8f;
typedef __attribute__((ext_vector_type(4)))  float    v4f;

#define WMMA_F16(A, B, C) \
  __builtin_amdgcn_wmma_f32_16x16x32_f16(false, (A), false, (B), (short)0, (C), false, false)

// ---- branch-free transcendentals on the HW TRANS pipe ----
__device__ __forceinline__ float fast_exp(float x) {
#if __has_builtin(__builtin_amdgcn_exp2f)
  return __builtin_amdgcn_exp2f(x * 1.44269504088896340736f);   // v_exp_f32
#else
  return __expf(x);
#endif
}
__device__ __forceinline__ float fast_rcp(float x) {
#if __has_builtin(__builtin_amdgcn_rcpf)
  return __builtin_amdgcn_rcpf(x);                              // v_rcp_f32
#else
  return 1.0f / x;
#endif
}
__device__ __forceinline__ float fast_sigmoid(float x) {
  return fast_rcp(1.0f + fast_exp(-x));
}
__device__ __forceinline__ float fast_tanh(float x) {
#if __has_builtin(__builtin_amdgcn_tanhf)
  return __builtin_amdgcn_tanhf(x);                             // v_tanh_f32 (CDNA5 HW op)
#else
  return 1.0f - 2.0f * fast_rcp(1.0f + fast_exp(2.0f * x));     // saturates correctly
#endif
}

// A-matrix fragment (16x32 f16, lane = M, documented CDNA5 layout):
// lane half=0: K = {0..7, 16..23}+kb ; half=1: +8.  Two contiguous 8-half runs.
__device__ __forceinline__ v16h fragA(const _Float16* p /* = row + kb + 8*half */) {
  v8h r0 = *(const v8h*)(p);
  v8h r1 = *(const v8h*)(p + 16);
  v16h f;
#pragma unroll
  for (int i = 0; i < 8; ++i) { f[i] = r0[i]; f[i + 8] = r1[i]; }
  return f;
}

// B-matrix fragment (32x16 f16, lane = N): same K order as A:
// lane half=0: K = {0..7, 16..23}+kb ; half=1: +8.
__device__ __forceinline__ v16h fragB(const _Float16* p /* = Wt row n + kb + 8*half */) {
  v8h r0 = *(const v8h*)(p);
  v8h r1 = *(const v8h*)(p + 16);
  v16h f;
#pragma unroll
  for (int i = 0; i < 8; ++i) { f[i] = r0[i]; f[i + 8] = r1[i]; }
  return f;
}

#define ATT_CHUNK  256   // timesteps of att staged per LDS refill (power of 2)
#define ATT_STRIDE 260   // row stride in floats (bank-skewed)

template <int CT>   // CT > 0: compile-time T (address folding); CT == 0: runtime T
__global__ __launch_bounds__(128) void agru_fused_kernel(
    const float* __restrict__ x,    const float* __restrict__ att,
    const float* __restrict__ h0,
    const float* __restrict__ w_ir, const float* __restrict__ w_hr,
    const float* __restrict__ b_ir, const float* __restrict__ b_hr,
    const float* __restrict__ w_ih, const float* __restrict__ w_hh,
    const float* __restrict__ b_ih, const float* __restrict__ b_hh,
    float* __restrict__ out, int T_rt)
{
  const int T = (CT > 0) ? CT : T_rt;

  __shared__ _Float16 Wt[4][64 * 64];       // transposed f16 weights, 32 KB
  __shared__ _Float16 xb[16 * 64];          // x tile (m-major), f16, 2 KB
  __shared__ _Float16 hb[16 * 64];          // h tile (m-major), f16, 2 KB
  __shared__ float    atb[16 * ATT_STRIDE]; // att chunk, ~16.6 KB

  const int tid  = threadIdx.x;
  const int wave = tid >> 5;
  const int lane = tid & 31;
  const int half = lane >> 4;          // K-half of this lane inside the wave
  const int ln   = lane & 15;
  const int n    = ln + 16 * wave;     // this lane's output column (C/D lane = N)
  const int b0   = blockIdx.x * 16;    // batch tile base

  // ---- one-time: transpose + convert the four 64x64 weights into LDS ----
  {
    const float* wsrc[4] = { w_ir, w_hr, w_ih, w_hh };
#pragma unroll
    for (int m = 0; m < 4; ++m) {
      const float* W = wsrc[m];
#pragma unroll 1
      for (int e = tid; e < 64 * 64; e += 128) {
        int k = e >> 6, c = e & 63;               // coalesced read of W[k][c]
        Wt[m][c * 64 + k] = (_Float16)W[k * 64 + c];
      }
    }
  }

  // per-lane bias scalars -> splat into WMMA C-init vectors (loop-invariant)
  v8f c_r, c_h, c_c;
  {
    const float br = b_ir[n] + b_hr[n];
    const float bh = b_ih[n];
    const float bc = b_hh[n];
#pragma unroll
    for (int j = 0; j < 8; ++j) { c_r[j] = br; c_h[j] = bh; c_c[j] = bc; }
  }

  // ---- init h: f32 master regs (C layout: VGPR j -> M = j + 8*half) ----
  v8f h;
#pragma unroll
  for (int j = 0; j < 8; ++j) {
    int m = j + 8 * half;
    float v = h0[(size_t)(b0 + m) * 64 + n];
    h[j] = v;
    hb[m * 64 + n] = (_Float16)v;
  }
  __syncthreads();   // Wt + hb ready

  // ---- loop-invariant B fragments: cached in VGPRs (8 x v16h = 64 VGPRs) ----
  const int boff = n * 64 + 8 * half;
  const v16h bir0 = fragB(&Wt[0][boff +  0]);
  const v16h bir1 = fragB(&Wt[0][boff + 32]);
  const v16h bhr0 = fragB(&Wt[1][boff +  0]);
  const v16h bhr1 = fragB(&Wt[1][boff + 32]);
  const v16h bih0 = fragB(&Wt[2][boff +  0]);
  const v16h bih1 = fragB(&Wt[2][boff + 32]);
  const v16h bhh0 = fragB(&Wt[3][boff +  0]);
  const v16h bhh1 = fragB(&Wt[3][boff + 32]);

  // x staging assignment: thread -> (row, 8-float column chunk)
  const int xr = tid >> 3;
  const int xc = (tid & 7) * 8;
  const float* xsrc = x + ((size_t)(b0 + xr) * T) * 64 + xc;

  // output base for this lane: row m = 8*half, col n; row j adds j*T*64
  // (compile-time for CT>0 -> folds into store IOFFSET).
  float* outp = out + ((size_t)(b0 + 8 * half) * T) * 64 + n;

  for (int t = 0; t < T; ++t) {
    // ---- refill att LDS chunk every ATT_CHUNK steps (once for T<=256) ----
    if ((t & (ATT_CHUNK - 1)) == 0) {
      if (t) __syncthreads();               // prior atb reads complete
      const int rem = (T - t < ATT_CHUNK) ? (T - t) : ATT_CHUNK;
#pragma unroll 1
      for (int m = 0; m < 16; ++m)
        for (int tt = tid; tt < rem; tt += 128)
          atb[m * ATT_STRIDE + tt] = att[(size_t)(b0 + m) * T + t + tt];
    }

    // ---- stage x[:, t, :] tile -> f16 LDS (x read exactly once from HBM) ----
    {
      const float* s = xsrc + (size_t)t * 64;
      v4f a = *(const v4f*)(s);
      v4f b = *(const v4f*)(s + 4);
      v8h o;
#pragma unroll
      for (int i = 0; i < 4; ++i) { o[i] = (_Float16)a[i]; o[i + 4] = (_Float16)b[i]; }
      *(v8h*)&xb[xr * 64 + xc] = o;
      __builtin_prefetch(s + 64, 0, 1);     // global_prefetch next step
    }
    __syncthreads();   // B1: xb/atb staged, hb(t) visible to all waves

    // ---- A fragments for this step (x tile + current h tile) ----
    const int aoff = 8 * half;
    const v16h ax0 = fragA(&xb[ln * 64 +  0 + aoff]);
    const v16h ax1 = fragA(&xb[ln * 64 + 32 + aoff]);
    const v16h ah0 = fragA(&hb[ln * 64 +  0 + aoff]);
    const v16h ah1 = fragA(&hb[ln * 64 + 32 + aoff]);

    // ---- 8 WMMAs; biases pre-loaded via the C operand of the first WMMA ----
    v8f accr = WMMA_F16(ax0, bir0, c_r);    // x@w_ir + h@w_hr + (b_ir+b_hr)
    accr     = WMMA_F16(ax1, bir1, accr);
    accr     = WMMA_F16(ah0, bhr0, accr);
    accr     = WMMA_F16(ah1, bhr1, accr);
    v8f acch = WMMA_F16(ax0, bih0, c_h);    // x@w_ih + b_ih
    acch     = WMMA_F16(ax1, bih1, acch);
    v8f accc = WMMA_F16(ah0, bhh0, c_c);    // h@w_hh + b_hh
    accc     = WMMA_F16(ah1, bhh1, accc);

    __syncthreads();   // B2: all frag reads done before xb/hb are overwritten

    // ---- elementwise gate math + attention-gated update (f32, branch-free) ----
    const int tc = t & (ATT_CHUNK - 1);
#pragma unroll
    for (int j = 0; j < 8; ++j) {
      int m = j + 8 * half;
      float at = atb[m * ATT_STRIDE + tc];                     // ds_load_b32
      float r  = fast_sigmoid(accr[j]);
      float hc = fast_tanh(acch[j] + r * accc[j]);
      float hn = h[j] + at * (hc - h[j]);
      h[j] = hn;
      outp[(size_t)j * T * 64 + (size_t)t * 64] = hn;          // imm-offset store
      hb[m * 64 + n] = (_Float16)hn;                           // f16 for next step
    }
  }
}

extern "C" void kernel_launch(void* const* d_in, const int* in_sizes, int n_in,
                              void* d_out, int out_size, void* d_ws, size_t ws_size,
                              hipStream_t stream) {
  const float* x    = (const float*)d_in[0];
  const float* att  = (const float*)d_in[1];
  const float* h0   = (const float*)d_in[2];
  const float* w_ir = (const float*)d_in[3];
  const float* w_hr = (const float*)d_in[4];
  const float* b_ir = (const float*)d_in[5];
  const float* b_hr = (const float*)d_in[6];
  const float* w_ih = (const float*)d_in[7];
  const float* w_hh = (const float*)d_in[8];
  const float* b_ih = (const float*)d_in[9];
  const float* b_hh = (const float*)d_in[10];

  const int U = in_sizes[5];           // 64  (b_ir)
  const int B = in_sizes[2] / U;       // 2048 (h0 = B*U)
  const int T = in_sizes[1] / B;       // 200  (att = B*T)

  dim3 grid(B / 16), block(128);
  if (T == 200) {
    agru_fused_kernel<200><<<grid, block, 0, stream>>>(
        x, att, h0, w_ir, w_hr, b_ir, b_hr, w_ih, w_hh, b_ih, b_hh,
        (float*)d_out, T);
  } else {
    agru_fused_kernel<0><<<grid, block, 0, stream>>>(
        x, att, h0, w_ir, w_hr, b_ir, b_hr, w_ih, w_hh, b_ih, b_hh,
        (float*)d_out, T);
  }
}
